// SpGAT_50225347559985
// MI455X (gfx1250) — matrix operand order repaired and checked by reading, not compiled, not run on hardware
//
#include <hip/hip_runtime.h>
#include <hip/hip_bf16.h>

// ---------------------------------------------------------------------------
// SpGAT forward for MI455X (gfx1250, wave32, WMMA + async-LDS staging).
// Dims: N=50000, F=R=128, NREL=500, E=150000, E2=30000, H=2.
// ---------------------------------------------------------------------------

typedef __attribute__((ext_vector_type(16))) __bf16 v16bf;
typedef __attribute__((ext_vector_type(8)))  float  v8f;
typedef __attribute__((ext_vector_type(4)))  int    v4i;

union ABu { unsigned int u[8]; uint4 q[2]; v16bf v; };
union Cu  { v8f v; float f[8]; };

__device__ __forceinline__ unsigned short f2bf(float f) {
    unsigned int u = __float_as_uint(f);
    unsigned int r = u + 0x7FFFu + ((u >> 16) & 1u);   // RNE
    return (unsigned short)(r >> 16);
}
__device__ __forceinline__ unsigned int pack2bf(float lo, float hi) {
    return (unsigned int)f2bf(lo) | ((unsigned int)f2bf(hi) << 16);
}
__device__ __forceinline__ float lrelu_neg_exp(float s) {
    float lr = s > 0.f ? s : 0.2f * s;                 // leaky_relu(s, 0.2)
    return __expf(-lr);                                // exp(-leaky_relu)
}
__device__ __forceinline__ float elu(float v) {
    return v > 0.f ? v : (__expf(v) - 1.f);
}

// --- CDNA5 async global->LDS 16B copy (per-lane gather), with safe fallback --
#if defined(__has_builtin)
#if __has_builtin(__builtin_amdgcn_global_load_async_to_lds_b128)
#define GAT_ASYNC_STAGING 1
#endif
#endif

__device__ __forceinline__ void copy16_g2l(unsigned short* lds, const unsigned short* g) {
#ifdef GAT_ASYNC_STAGING
    __builtin_amdgcn_global_load_async_to_lds_b128(
        (__attribute__((address_space(1))) v4i*)g,     // C-style cast drops const
        (__attribute__((address_space(3))) v4i*)lds, 0, 0);
#else
    *(uint4*)lds = *(const uint4*)g;
#endif
}
__device__ __forceinline__ void stage_fence() {
#ifdef GAT_ASYNC_STAGING
#if defined(__has_builtin) && __has_builtin(__builtin_amdgcn_s_wait_asynccnt)
    __builtin_amdgcn_s_wait_asynccnt(0);
#else
    asm volatile("s_wait_asynccnt 0" ::: "memory");
#endif
#endif
}

// ---------------------------------------------------------------------------
// Utility kernels
// ---------------------------------------------------------------------------
__global__ void k_zero_f32(float* p, long long n) {
    long long i = (long long)blockIdx.x * blockDim.x + threadIdx.x;
    long long st = (long long)gridDim.x * blockDim.x;
    for (; i < n; i += st) p[i] = 0.f;
}

// fp32 -> bf16, 8 elements per thread (n must be a multiple of 8; it is).
__global__ void k_cvt_bf16x8(const float* __restrict__ src,
                             unsigned int* __restrict__ dst, long long n8) {
    long long i = (long long)blockIdx.x * blockDim.x + threadIdx.x;
    long long st = (long long)gridDim.x * blockDim.x;
    for (; i < n8; i += st) {
        const float4* s = (const float4*)(src + i * 8);
        float4 a = s[0], b = s[1];
        uint4 q;
        q.x = pack2bf(a.x, a.y); q.y = pack2bf(a.z, a.w);
        q.z = pack2bf(b.x, b.y); q.w = pack2bf(b.z, b.w);
        ((uint4*)dst)[i] = q;
    }
}

// Pack a [2,128,384] into WMMA-B fragment-linear order:
// bfrag1[h][kt=12][dt=8][lane=32][v=8] : u = pack(a[h][dt*16+col][K], K+1),
// K = kt*32 + 16*(lane>>4) + 2v.
__global__ void k_pack_b1(const float* __restrict__ a, unsigned int* __restrict__ bf) {
    int i = blockIdx.x * blockDim.x + threadIdx.x;
    if (i >= 2 * 12 * 8 * 32 * 8) return;
    int r = i;
    int v = r & 7; r >>= 3;
    int lane = r & 31; r >>= 5;
    int dt = r & 7; r >>= 3;
    int kt = r % 12;
    int h = r / 12;
    int K = kt * 32 + ((lane >> 4) << 4) + 2 * v;
    int d = dt * 16 + (lane & 15);
    const float* row = a + (long long)(h * 128 + d) * 384 + K;
    bf[i] = pack2bf(row[0], row[1]);
}

// Pack a_out [256,768] into bfrag2[kt=24][dt=16][lane=32][v=8].
__global__ void k_pack_b2(const float* __restrict__ a, unsigned int* __restrict__ bf) {
    int i = blockIdx.x * blockDim.x + threadIdx.x;
    if (i >= 24 * 16 * 32 * 8) return;
    int r = i;
    int v = r & 7; r >>= 3;
    int lane = r & 31; r >>= 5;
    int dt = r & 15; r >>= 4;
    int kt = r;                                   // 0..23
    int K = kt * 32 + ((lane >> 4) << 4) + 2 * v;
    int d = dt * 16 + (lane & 15);
    const float* row = a + (long long)d * 768 + K;
    bf[i] = pack2bf(row[0], row[1]);
}

// ---------------------------------------------------------------------------
// out_relation_1 = relation_embed [NREL,128] @ W [128,256] -> f32 [NREL,256]
// One wave per 16x16 tile; branchless (clamped row + mask), vector A loads.
// ---------------------------------------------------------------------------
__global__ void k_outrel_gemm(const float* __restrict__ rel,
                              const float* __restrict__ W,
                              float* __restrict__ out, int NREL) {
    int tile = blockIdx.x;
    int tm = tile >> 4, tn = tile & 15;            // 16 N-tiles (D=256)
    int lane = threadIdx.x & 31;
    int hl = lane >> 4, col = lane & 15;
    int mBase = tm * 16, dBase = tn * 16;

    Cu c;
    #pragma unroll
    for (int v = 0; v < 8; v++) c.f[v] = 0.f;

    int m = mBase + col;
    float msk = (m < NREL) ? 1.f : 0.f;
    int mc = m < NREL ? m : (NREL - 1);
    const float* arow = rel + (long long)mc * 128;
    int d = dBase + col;

    for (int kb = 0; kb < 128; kb += 32) {
        ABu A, B;
        const float4* a0 = (const float4*)(arow + kb + hl * 16);
        const float4* a1 = (const float4*)(arow + kb + hl * 16 + 8);
        float4 f0 = a0[0], f1 = a0[1], f2 = a1[0], f3 = a1[1];
        A.u[0] = pack2bf(f0.x * msk, f0.y * msk);
        A.u[1] = pack2bf(f0.z * msk, f0.w * msk);
        A.u[2] = pack2bf(f1.x * msk, f1.y * msk);
        A.u[3] = pack2bf(f1.z * msk, f1.w * msk);
        A.u[4] = pack2bf(f2.x * msk, f2.y * msk);
        A.u[5] = pack2bf(f2.z * msk, f2.w * msk);
        A.u[6] = pack2bf(f3.x * msk, f3.y * msk);
        A.u[7] = pack2bf(f3.z * msk, f3.w * msk);
        #pragma unroll
        for (int v = 0; v < 8; v++) {
            int kB = kb + hl * 16 + v * 2;
            B.u[v] = pack2bf(W[(long long)kB * 256 + d], W[(long long)(kB + 1) * 256 + d]);
        }
        c.v = __builtin_amdgcn_wmma_f32_16x16x32_bf16(false, A.v, false, B.v,
                                                      (short)0, c.v, false, false);
    }
    #pragma unroll
    for (int v = 0; v < 8; v++) {
        int row = mBase + v + 8 * hl;
        if (row < NREL) out[(long long)row * 256 + dBase + col] = c.f[v];
    }
}

// ---------------------------------------------------------------------------
// Layer-1 fused edge kernel. Grid: (numTiles, H=2), block = 256 (8 waves).
// 16 edges/block; LDS-staged A [16 x 384] bf16 via async b128 gathers; each
// wave computes one 16x16 D-tile (D=128) with 12 WMMAs (A: 2x ds_load_b128,
// B: 2x global b128 from fragment-linear weights); scores reduced via LDS
// atomics; weighted rows scattered with global float atomics.
// ---------------------------------------------------------------------------
__global__ void __launch_bounds__(256)
k_gat1(const unsigned short* __restrict__ xbf,     // [N,128] bf16
       const float* __restrict__ eemb,             // [E,128]
       const float* __restrict__ rel,              // [NREL,128]
       const int* __restrict__ el,                 // [2,E]
       const int* __restrict__ eln,                // [2,E2]
       const int* __restrict__ etn,                // [E2,2]
       const unsigned int* __restrict__ bfrag1,    // [2][12][8][32][8]
       const float* __restrict__ a2,               // [2,128]
       float* __restrict__ hp1,                    // [2][N][128] accum
       float* __restrict__ rs1,                    // [2][N] accum
       int N, int E, int E2) {
    __shared__ __align__(16) unsigned short As[16 * 392];  // row stride 392 (784B, 16B-aligned, bank-clean)
    __shared__ float score[16];
    __shared__ int sIdx[16], dIdx[16], t0s[16], t1s[16];

    const int ET = E + E2;
    const int tile = blockIdx.x, head = blockIdx.y;
    const int e0 = tile * 16;
    const int tid = threadIdx.x;

    if (tid < 16) {
        int e = e0 + tid;
        int s = 0, d = 0, u0 = 0, u1 = 0;
        if (e < ET) {
            if (e < E) { s = el[e]; d = el[E + e]; }
            else { int j = e - E; s = eln[j]; d = eln[E2 + j]; u0 = etn[2 * j]; u1 = etn[2 * j + 1]; }
        }
        sIdx[tid] = s; dIdx[tid] = d; t0s[tid] = u0; t1s[tid] = u1;
        score[tid] = 0.f;
    }
    __syncthreads();

    // ---- stage gathered tile: [x[src] | x[dst] | edge_feat] (bf16) ----
    {
        const int m = tid >> 4, part = tid & 15;   // 16B chunk per thread
        const int e = e0 + m;
        const bool ev = (e < ET);
        uint4 z = make_uint4(0u, 0u, 0u, 0u);
        unsigned short* l0 = &As[m * 392 + part * 8];
        unsigned short* l1 = &As[m * 392 + 128 + part * 8];
        if (ev) {
            copy16_g2l(l0, xbf + (long long)sIdx[m] * 128 + part * 8);
            copy16_g2l(l1, xbf + (long long)dIdx[m] * 128 + part * 8);
        } else {
            *(uint4*)l0 = z;
            *(uint4*)l1 = z;
        }
        // edge-feature region: 8 fp32 -> bf16 per thread
        unsigned short* l2 = &As[m * 392 + 256 + part * 8];
        if (ev) {
            float f[8];
            if (e < E) {
                const float4* p = (const float4*)(eemb + (long long)e * 128 + part * 8);
                float4 x0 = p[0], x1 = p[1];
                f[0]=x0.x; f[1]=x0.y; f[2]=x0.z; f[3]=x0.w;
                f[4]=x1.x; f[5]=x1.y; f[6]=x1.z; f[7]=x1.w;
            } else {
                const float4* p0 = (const float4*)(rel + (long long)t0s[m] * 128 + part * 8);
                const float4* p1 = (const float4*)(rel + (long long)t1s[m] * 128 + part * 8);
                float4 a0 = p0[0], a1 = p0[1], b0 = p1[0], b1 = p1[1];
                f[0]=a0.x+b0.x; f[1]=a0.y+b0.y; f[2]=a0.z+b0.z; f[3]=a0.w+b0.w;
                f[4]=a1.x+b1.x; f[5]=a1.y+b1.y; f[6]=a1.z+b1.z; f[7]=a1.w+b1.w;
            }
            uint4 q;
            q.x = pack2bf(f[0], f[1]); q.y = pack2bf(f[2], f[3]);
            q.z = pack2bf(f[4], f[5]); q.w = pack2bf(f[6], f[7]);
            *(uint4*)l2 = q;
        } else {
            *(uint4*)l2 = z;
        }
    }
    stage_fence();
    __syncthreads();

    const int wave = tid >> 5, lane = tid & 31;
    const int hl = lane >> 4, col = lane & 15;
    const int dBase = wave * 16;
    const unsigned int* Bh = bfrag1 + (long long)head * (12 * 8 * 32 * 8);

    Cu c;
    #pragma unroll
    for (int v = 0; v < 8; v++) c.f[v] = 0.f;

    const unsigned short* arow = &As[col * 392];
    for (int kt = 0; kt < 12; ++kt) {
        ABu A, B;
        int kb = kt * 32;
        A.q[0] = *(const uint4*)(arow + kb + hl * 16);       // ds_load_b128
        A.q[1] = *(const uint4*)(arow + kb + hl * 16 + 8);   // ds_load_b128
        const uint4* bq = (const uint4*)(Bh + ((((kt * 8 + wave) * 32) + lane) << 3));
        B.q[0] = bq[0];                                      // global b128
        B.q[1] = bq[1];                                      // global b128
        c.v = __builtin_amdgcn_wmma_f32_16x16x32_bf16(false, A.v, false, B.v,
                                                      (short)0, c.v, false, false);
    }

    // score partial: edge_m[row] . a2 (cross-wave LDS reduction)
    float a2d = a2[head * 128 + dBase + col];
    #pragma unroll
    for (int v = 0; v < 8; v++) atomicAdd(&score[v + 8 * hl], c.f[v] * a2d);
    __syncthreads();

    if (tid < 16) {
        int e = e0 + tid;
        if (e < ET) atomicAdd(&rs1[(long long)head * N + sIdx[tid]], lrelu_neg_exp(score[tid]));
    }
    float* hp = hp1 + (long long)head * N * 128;
    #pragma unroll
    for (int v = 0; v < 8; v++) {
        int m = v + 8 * hl;
        int e = e0 + m;
        if (e < ET) {
            float w = lrelu_neg_exp(score[m]);
            atomicAdd(&hp[(long long)sIdx[m] * 128 + dBase + col], w * c.f[v]);
        }
    }
}

// Layer-1 finalize: x2_bf[n][h*128+d] = bf16(elu(hp1/rowsum))
__global__ void k_fin1(const float* __restrict__ hp1, const float* __restrict__ rs1,
                       unsigned short* __restrict__ x2bf, int N) {
    long long total = (long long)N * 256;
    long long i = (long long)blockIdx.x * blockDim.x + threadIdx.x;
    long long st = (long long)gridDim.x * blockDim.x;
    for (; i < total; i += st) {
        int n = (int)(i >> 8), j = (int)(i & 255);
        int h = j >> 7, d = j & 127;
        float rs = rs1[(long long)h * N + n];
        rs = (rs == 0.f) ? 1e-12f : rs;
        float v = hp1[((long long)h * N + n) * 128 + d] / rs;
        x2bf[i] = f2bf(elu(v));
    }
}

// ---------------------------------------------------------------------------
// Layer-2 fused edge kernel. Grid: numTiles, block = 512 (16 waves).
// K = 768 (x2[src] | x2[dst] | out_relation_1[type]), D = 256.
// ---------------------------------------------------------------------------
__global__ void __launch_bounds__(512)
k_gat2(const unsigned short* __restrict__ x2bf,    // [N,256] bf16
       const float* __restrict__ orel,             // [NREL,256] f32 (in d_out)
       const int* __restrict__ el, const int* __restrict__ eln,
       const int* __restrict__ et, const int* __restrict__ etn,
       const unsigned int* __restrict__ bfrag2,    // [24][16][32][8]
       const float* __restrict__ a2o,              // [256]
       float* __restrict__ outx,                   // [N,256] accum (d_out)
       float* __restrict__ rs2,                    // [N]
       int N, int E, int E2) {
    __shared__ __align__(16) unsigned short As[16 * 776];  // row stride 776 (1552B, bank-clean)
    __shared__ float score[16];
    __shared__ int sIdx[16], dIdx[16], ty0[16], ty1[16];

    const int ET = E + E2;
    const int tile = blockIdx.x;
    const int e0 = tile * 16;
    const int tid = threadIdx.x;

    if (tid < 16) {
        int e = e0 + tid;
        int s = 0, d = 0, u0 = 0, u1 = -1;        // u1 < 0 => 1-hop edge
        if (e < ET) {
            if (e < E) { s = el[e]; d = el[E + e]; u0 = et[e]; }
            else { int j = e - E; s = eln[j]; d = eln[E2 + j]; u0 = etn[2 * j]; u1 = etn[2 * j + 1]; }
        }
        sIdx[tid] = s; dIdx[tid] = d; ty0[tid] = u0; ty1[tid] = u1;
        score[tid] = 0.f;
    }
    __syncthreads();

    // ---- stage tile ----
    {
        const int m = tid >> 5, part = tid & 31;   // 16B chunk per thread
        const int e = e0 + m;
        const bool ev = (e < ET);
        uint4 z = make_uint4(0u, 0u, 0u, 0u);
        unsigned short* l0 = &As[m * 776 + part * 8];
        unsigned short* l1 = &As[m * 776 + 256 + part * 8];
        if (ev) {
            copy16_g2l(l0, x2bf + (long long)sIdx[m] * 256 + part * 8);
            copy16_g2l(l1, x2bf + (long long)dIdx[m] * 256 + part * 8);
        } else {
            *(uint4*)l0 = z;
            *(uint4*)l1 = z;
        }
        unsigned short* l2 = &As[m * 776 + 512 + part * 8];
        if (ev) {
            const float4* p0 = (const float4*)(orel + (long long)ty0[m] * 256 + part * 8);
            float4 a0 = p0[0], a1 = p0[1];
            if (ty1[m] >= 0) {
                const float4* p1 = (const float4*)(orel + (long long)ty1[m] * 256 + part * 8);
                float4 b0 = p1[0], b1 = p1[1];
                a0.x+=b0.x; a0.y+=b0.y; a0.z+=b0.z; a0.w+=b0.w;
                a1.x+=b1.x; a1.y+=b1.y; a1.z+=b1.z; a1.w+=b1.w;
            }
            uint4 q;
            q.x = pack2bf(a0.x, a0.y); q.y = pack2bf(a0.z, a0.w);
            q.z = pack2bf(a1.x, a1.y); q.w = pack2bf(a1.z, a1.w);
            *(uint4*)l2 = q;
        } else {
            *(uint4*)l2 = z;
        }
    }
    stage_fence();
    __syncthreads();

    const int wave = tid >> 5, lane = tid & 31;
    const int hl = lane >> 4, col = lane & 15;
    const int dBase = wave * 16;

    Cu c;
    #pragma unroll
    for (int v = 0; v < 8; v++) c.f[v] = 0.f;

    const unsigned short* arow = &As[col * 776];
    for (int kt = 0; kt < 24; ++kt) {
        ABu A, B;
        int kb = kt * 32;
        A.q[0] = *(const uint4*)(arow + kb + hl * 16);
        A.q[1] = *(const uint4*)(arow + kb + hl * 16 + 8);
        const uint4* bq = (const uint4*)(bfrag2 + ((((kt * 16 + wave) * 32) + lane) << 3));
        B.q[0] = bq[0];
        B.q[1] = bq[1];
        c.v = __builtin_amdgcn_wmma_f32_16x16x32_bf16(false, A.v, false, B.v,
                                                      (short)0, c.v, false, false);
    }

    float a2d = a2o[dBase + col];
    #pragma unroll
    for (int v = 0; v < 8; v++) atomicAdd(&score[v + 8 * hl], c.f[v] * a2d);
    __syncthreads();

    if (tid < 16) {
        int e = e0 + tid;
        if (e < ET) atomicAdd(&rs2[sIdx[tid]], lrelu_neg_exp(score[tid]));
    }
    #pragma unroll
    for (int v = 0; v < 8; v++) {
        int m = v + 8 * hl;
        int e = e0 + m;
        if (e < ET) {
            float w = lrelu_neg_exp(score[m]);
            atomicAdd(&outx[(long long)sIdx[m] * 256 + dBase + col], w * c.f[v]);
        }
    }
}

// Layer-2 finalize (in place on d_out): elu(h'/rowsum)
__global__ void k_fin2(float* __restrict__ outx, const float* __restrict__ rs2, int N) {
    long long total = (long long)N * 256;
    long long i = (long long)blockIdx.x * blockDim.x + threadIdx.x;
    long long st = (long long)gridDim.x * blockDim.x;
    for (; i < total; i += st) {
        int n = (int)(i >> 8);
        float rs = rs2[n];
        rs = (rs == 0.f) ? 1e-12f : rs;
        outx[i] = elu(outx[i] / rs);
    }
}

// ---------------------------------------------------------------------------
extern "C" void kernel_launch(void* const* d_in, const int* in_sizes, int n_in,
                              void* d_out, int out_size, void* d_ws, size_t ws_size,
                              hipStream_t stream) {
    (void)n_in; (void)out_size; (void)ws_size;
    const float* ent   = (const float*)d_in[0];   // [N,128]
    const float* rel   = (const float*)d_in[1];   // [NREL,128]
    const float* eemb  = (const float*)d_in[2];   // [E,128]
    const int*   el    = (const int*)d_in[3];     // [2,E]
    const int*   et    = (const int*)d_in[4];     // [E]
    const int*   eln   = (const int*)d_in[5];     // [2,E2]
    const int*   etn   = (const int*)d_in[6];     // [E2,2]
    const float* a     = (const float*)d_in[7];   // [2,128,384]
    const float* a2    = (const float*)d_in[8];   // [2,128]
    const float* W     = (const float*)d_in[9];   // [128,256]
    const float* aout  = (const float*)d_in[10];  // [256,768]
    const float* a2o   = (const float*)d_in[11];  // [256]

    const int N    = in_sizes[0] / 128;
    const int NREL = in_sizes[1] / 128;
    const int E    = in_sizes[4];
    const int E2   = in_sizes[6] / 2;
    const int ET   = E + E2;
    const int tiles = (ET + 15) / 16;

    // Workspace carve
    char* wsp = (char*)d_ws;
    size_t off = 0;
    auto carve = [&](size_t bytes) -> char* {
        char* p = wsp + off;
        off += (bytes + 255) & ~(size_t)255;
        return p;
    };
    unsigned short* xbf  = (unsigned short*)carve((size_t)N * 128 * 2);
    unsigned short* x2bf = (unsigned short*)carve((size_t)N * 256 * 2);
    float* hp1           = (float*)carve((size_t)2 * N * 128 * 4);
    float* rs1           = (float*)carve((size_t)2 * N * 4);
    float* rs2           = (float*)carve((size_t)N * 4);
    unsigned int* bfrag1 = (unsigned int*)carve((size_t)2 * 12 * 8 * 32 * 8 * 4);
    unsigned int* bfrag2 = (unsigned int*)carve((size_t)24 * 16 * 32 * 8 * 4);

    float* outX = (float*)d_out;                  // [N,256]
    float* outR = outX + (long long)N * 256;      // [NREL,256]

    // 0) zero accumulators (atomics) — every call, graph-capture safe
    k_zero_f32<<<2048, 256, 0, stream>>>(hp1, (long long)2 * N * 128);
    k_zero_f32<<<256, 256, 0, stream>>>(rs1, (long long)2 * N);
    k_zero_f32<<<256, 256, 0, stream>>>(rs2, (long long)N);
    k_zero_f32<<<2048, 256, 0, stream>>>(outX, (long long)N * 256);

    // 1) prep: bf16 entity table + fragment-linear WMMA B operands
    k_cvt_bf16x8<<<2048, 256, 0, stream>>>(ent, (unsigned int*)xbf, (long long)N * 128 / 8);
    k_pack_b1<<<(2 * 12 * 8 * 32 * 8 + 255) / 256, 256, 0, stream>>>(a, bfrag1);
    k_pack_b2<<<(24 * 16 * 32 * 8 + 255) / 256, 256, 0, stream>>>(aout, bfrag2);

    // 2) out_relation_1 = relation_embed @ W  (WMMA, straight into d_out)
    {
        int tilesM = (NREL + 15) / 16;
        k_outrel_gemm<<<tilesM * 16, 32, 0, stream>>>(rel, W, outR, NREL);
    }

    // 3) layer-1 fused edge GEMM + softmax-weight scatter (per head)
    k_gat1<<<dim3(tiles, 2), 256, 0, stream>>>(xbf, eemb, rel, el, eln, etn,
                                               bfrag1, a2, hp1, rs1, N, E, E2);

    // 4) normalize + ELU -> bf16 x2 [N,256]
    k_fin1<<<2048, 256, 0, stream>>>(hp1, rs1, x2bf, N);

    // 5) layer-2 fused edge GEMM + scatter into d_out
    k_gat2<<<tiles, 512, 0, stream>>>(x2bf, outR, el, eln, et, etn,
                                      bfrag2, a2o, outX, rs2, N, E, E2);

    // 6) normalize + ELU in place
    k_fin2<<<2048, 256, 0, stream>>>(outX, rs2, N);
}
